// TernaryLinear_12206297055408
// MI455X (gfx1250) — compile-verified, matrix operand order checked by reading, not run on hardware
//
#include <hip/hip_runtime.h>

// ---------------------------------------------------------------------------
// TernaryLinear for MI455X (gfx1250, wave32, WMMA)
//   out[b,s,o] = alpha * (x @ ternarize(w)^T) + bias
//   M = B*S = 8192, N = D_OUT = 8192, K = D_IN = 2048
// bf16 WMMA GEMM (ternary weights exact in bf16), f32 accumulation.
// Block tile 128x256, BK=32; 8 waves as 2(M)x4(N), wave tile 64x64
// -> 16 v_wmma_f32_16x16x32_bf16 per wave per k-step.
// Data movement: GLOBAL_LOAD_ASYNC_TO_LDS_B128 (ASYNCcnt), double-buffered
// LDS selected by pointer arithmetic; k-loop forced to unroll(1) so the
// in-place WMMA accumulators coalesce (no v_mov walls, <256 VGPRs).
// ---------------------------------------------------------------------------

typedef __attribute__((ext_vector_type(16))) __bf16 v16bf;
typedef __attribute__((ext_vector_type(8)))  float  v8f;

constexpr int    MDIM = 4 * 2048;   // B*S
constexpr int    NDIM = 8192;       // D_OUT
constexpr int    KDIM = 2048;       // D_IN
constexpr size_t WELEMS = (size_t)NDIM * KDIM;   // 16,777,216
constexpr size_t XELEMS = (size_t)MDIM * KDIM;   // 16,777,216

#if defined(__HIP_DEVICE_COMPILE__) && \
    __has_builtin(__builtin_amdgcn_global_load_async_to_lds_b128)
#define USE_ASYNC_LDS 1
#else
#define USE_ASYNC_LDS 0
#endif

#if defined(__HIP_DEVICE_COMPILE__) && \
    __has_builtin(__builtin_amdgcn_s_wait_asynccnt)
#define WAIT_ASYNC(n) __builtin_amdgcn_s_wait_asynccnt(n)
#elif defined(__HIP_DEVICE_COMPILE__)
#define WAIT_ASYNC(n) asm volatile("s_wait_asynccnt " #n ::: "memory")
#else
#define WAIT_ASYNC(n) ((void)0)
#endif

#if USE_ASYNC_LDS
// Builtin signature (from clang): (v4i addrspace(1)*, v4i addrspace(3)*, Ii, Ii)
typedef int v4i_vs __attribute__((vector_size(16)));
typedef __attribute__((address_space(1))) v4i_vs v4i_g;   // global
typedef __attribute__((address_space(3))) v4i_vs v4i_l;   // LDS

__device__ __forceinline__ void async_cp16(const unsigned short* g, unsigned short* l) {
    __builtin_amdgcn_global_load_async_to_lds_b128((v4i_g*)g, (v4i_l*)l, 0, 0);
}
#endif

// ---------------- pass 1: per-block sum of |w| (fixed-order, deterministic)
__global__ __launch_bounds__(256) void reduce_abs_sum(const float* __restrict__ w,
                                                      float* __restrict__ partials) {
    __shared__ float s[256];
    const int t = threadIdx.x;
    const size_t base = (size_t)blockIdx.x * 4096;
    float acc = 0.f;
#pragma unroll
    for (int i = 0; i < 16; ++i) acc += fabsf(w[base + t + i * 256]);
    s[t] = acc;
    __syncthreads();
    for (int off = 128; off > 0; off >>= 1) {
        if (t < off) s[t] += s[t + off];
        __syncthreads();
    }
    if (t == 0) partials[blockIdx.x] = s[0];
}

// ---------------- pass 2: reduce 4096 partials -> delta = 0.7*mean(|w|)
__global__ __launch_bounds__(256) void finalize_delta(const float* __restrict__ partials,
                                                      float* __restrict__ delta) {
    __shared__ float s[256];
    const int t = threadIdx.x;
    float acc = 0.f;
#pragma unroll
    for (int i = 0; i < 16; ++i) acc += partials[t + i * 256];
    s[t] = acc;
    __syncthreads();
    for (int off = 128; off > 0; off >>= 1) {
        if (t < off) s[t] += s[t + off];
        __syncthreads();
    }
    if (t == 0) *delta = 0.7f * s[0] * (1.0f / (float)WELEMS);
}

// ---------------- ternarize weight -> bf16 {-1, 0, +1}
__global__ __launch_bounds__(256) void quantize_w(const float* __restrict__ w,
                                                  const float* __restrict__ dptr,
                                                  unsigned short* __restrict__ wq) {
    const float d = *dptr;
    const size_t i = ((size_t)blockIdx.x * 256 + threadIdx.x) * 4;
    const float4 v = *(const float4*)(w + i);
    const float f[4] = {v.x, v.y, v.z, v.w};
    unsigned int q[4];
#pragma unroll
    for (int j = 0; j < 4; ++j)
        q[j] = (f[j] > d) ? 0x3F80u : ((f[j] < -d) ? 0xBF80u : 0u);
    uint2 p;
    p.x = q[0] | (q[1] << 16);
    p.y = q[2] | (q[3] << 16);
    *(uint2*)(wq + i) = p;
}

// ---------------- fp32 -> bf16 (round-to-nearest-even)
__device__ __forceinline__ unsigned int f32_to_bf16_bits(float f) {
    unsigned int u = __float_as_uint(f);
    u += 0x7FFFu + ((u >> 16) & 1u);
    return u >> 16;
}

__global__ __launch_bounds__(256) void convert_x(const float* __restrict__ x,
                                                 unsigned short* __restrict__ xb) {
    const size_t i = ((size_t)blockIdx.x * 256 + threadIdx.x) * 4;
    const float4 v = *(const float4*)(x + i);
    uint2 p;
    p.x = f32_to_bf16_bits(v.x) | (f32_to_bf16_bits(v.y) << 16);
    p.y = f32_to_bf16_bits(v.z) | (f32_to_bf16_bits(v.w) << 16);
    *(uint2*)(xb + i) = p;
}

// ---------------- WMMA GEMM --------------------------------------------------
union FragAB {
    v16bf v;
    uint4 q[2];
};

__global__ __launch_bounds__(256) void ternary_gemm_bf16(
    const unsigned short* __restrict__ Abf,   // [M,K] bf16 bits (x)
    const unsigned short* __restrict__ Bbf,   // [N,K] bf16 bits (w ternary)
    const float* __restrict__ alpha_p,
    const float* __restrict__ bias,
    float* __restrict__ out) {
    constexpr int BM = 128, BN = 256, BK = 32, LDT = BK + 8;   // LDT in bf16
    constexpr int KSTEPS = KDIM / BK;                          // 64
    constexpr int ASZ = BM * LDT;                              // elems / buffer
    constexpr int BSZ = BN * LDT;

    __shared__ unsigned short sA[2 * ASZ];   // 2 x 10240 B
    __shared__ unsigned short sB[2 * BSZ];   // 2 x 20480 B

    const int t = threadIdx.x;
    const int mBase = blockIdx.y * BM;
    const int nBase = blockIdx.x * BN;

    // ---- cooperative loader mapping
    const int arow = t >> 1;          // A: 128 rows x 2 segs of 16 bf16
    const int aseg = (t & 1) * 16;
    const int brow = t;               // B: 256 rows, 32 bf16 each (4 x b128)

    // ---- wave/frag mapping (ISA 16-bit A layout & 32-bit C layout)
    const int wave   = t >> 5;
    const int lane   = t & 31;
    const int lane16 = lane & 15;
    const int laneHi = lane >> 4;
    const int kSel   = laneHi * 8;          // K base within BK slab
    const int mWave  = (wave >> 2) * 64;    // 2 waves along M
    const int nWave  = (wave & 3) * 64;     // 4 waves along N

    v8f zero = {};
    v8f acc[16];
#pragma unroll
    for (int i = 0; i < 16; ++i) acc[i] = zero;

    const unsigned short* gA = Abf + (size_t)(mBase + arow) * KDIM + aseg;
    const unsigned short* gB = Bbf + (size_t)(nBase + brow) * KDIM;

    // one k-step of matrix math from LDS buffers (bA, bB).
    // A-frags stay resident; B-frags stream one at a time (low live range).
    auto compute_tile = [&](const unsigned short* bA, const unsigned short* bB) {
        FragAB a[4];
#pragma unroll
        for (int mi = 0; mi < 4; ++mi) {
            const int row = mWave + mi * 16 + lane16;
            a[mi].q[0] = *(const uint4*)(bA + row * LDT + kSel);
            a[mi].q[1] = *(const uint4*)(bA + row * LDT + kSel + 16);
        }
#pragma unroll
        for (int ni = 0; ni < 4; ++ni) {
            const int col = nWave + ni * 16 + lane16;
            FragAB b;
            b.q[0] = *(const uint4*)(bB + col * LDT + kSel);
            b.q[1] = *(const uint4*)(bB + col * LDT + kSel + 16);
#pragma unroll
            for (int mi = 0; mi < 4; ++mi)
                acc[mi * 4 + ni] = __builtin_amdgcn_wmma_f32_16x16x32_bf16(
                    false, a[mi].v, false, b.v, (short)0, acc[mi * 4 + ni],
                    false, false);
        }
    };

    unsigned short* baseA = &sA[0];
    unsigned short* baseB = &sB[0];

#if USE_ASYNC_LDS
    // ---- CDNA5 async copy path: GLOBAL_LOAD_ASYNC_TO_LDS_B128, ASYNCcnt.
    // 6 async ops per thread per tile; completion is in-order, so
    // asynccnt <= 6 proves the previous tile fully landed.
    auto issue_tile = [&](int kt, unsigned short* lA, unsigned short* lB) {
        const unsigned short* pA = gA + (size_t)kt * BK;
        const unsigned short* pB = gB + (size_t)kt * BK;
        async_cp16(pA + 0, lA + arow * LDT + aseg + 0);
        async_cp16(pA + 8, lA + arow * LDT + aseg + 8);
        async_cp16(pB + 0,  lB + brow * LDT + 0);
        async_cp16(pB + 8,  lB + brow * LDT + 8);
        async_cp16(pB + 16, lB + brow * LDT + 16);
        async_cp16(pB + 24, lB + brow * LDT + 24);
    };

    issue_tile(0, baseA, baseB);
#pragma unroll 1
    for (int ks = 0; ks < KSTEPS - 1; ++ks) {
        const int cur = (ks & 1);
        const int nxt = cur ^ 1;
        issue_tile(ks + 1, baseA + nxt * ASZ, baseB + nxt * BSZ);  // buffer idle
        WAIT_ASYNC(6);                 // my tile-ks writes complete (in-order)
        __syncthreads();               // everyone's tile-ks writes complete
        compute_tile(baseA + cur * ASZ, baseB + cur * BSZ);
        __syncthreads();               // all reads done before ks+2 overwrites
    }
    WAIT_ASYNC(0);
    __syncthreads();
    {
        const int cur = (KSTEPS - 1) & 1;
        compute_tile(baseA + cur * ASZ, baseB + cur * BSZ);
    }
#else
    // ---- fallback: global->reg staging, double buffer, 1 barrier per step.
    uint4 ra0, ra1, rb0, rb1, rb2, rb3;
    ra0 = *(const uint4*)(gA + 0);
    ra1 = *(const uint4*)(gA + 8);
    rb0 = *(const uint4*)(gB + 0);
    rb1 = *(const uint4*)(gB + 8);
    rb2 = *(const uint4*)(gB + 16);
    rb3 = *(const uint4*)(gB + 24);
    *(uint4*)(baseA + arow * LDT + aseg + 0) = ra0;
    *(uint4*)(baseA + arow * LDT + aseg + 8) = ra1;
    *(uint4*)(baseB + brow * LDT + 0)  = rb0;
    *(uint4*)(baseB + brow * LDT + 8)  = rb1;
    *(uint4*)(baseB + brow * LDT + 16) = rb2;
    *(uint4*)(baseB + brow * LDT + 24) = rb3;
    __syncthreads();

#pragma unroll 1
    for (int ks = 0; ks < KSTEPS - 1; ++ks) {   // branch-free body
        const unsigned short* pA = gA + (size_t)(ks + 1) * BK;
        const unsigned short* pB = gB + (size_t)(ks + 1) * BK;
        ra0 = *(const uint4*)(pA + 0);
        ra1 = *(const uint4*)(pA + 8);
        rb0 = *(const uint4*)(pB + 0);
        rb1 = *(const uint4*)(pB + 8);
        rb2 = *(const uint4*)(pB + 16);
        rb3 = *(const uint4*)(pB + 24);

        const int cur = (ks & 1);
        compute_tile(baseA + cur * ASZ, baseB + cur * BSZ);

        unsigned short* nA = baseA + (cur ^ 1) * ASZ;   // idle since ks-1 barrier
        unsigned short* nB = baseB + (cur ^ 1) * BSZ;
        *(uint4*)(nA + arow * LDT + aseg + 0) = ra0;
        *(uint4*)(nA + arow * LDT + aseg + 8) = ra1;
        *(uint4*)(nB + brow * LDT + 0)  = rb0;
        *(uint4*)(nB + brow * LDT + 8)  = rb1;
        *(uint4*)(nB + brow * LDT + 16) = rb2;
        *(uint4*)(nB + brow * LDT + 24) = rb3;
        __syncthreads();
    }
    {
        const int cur = (KSTEPS - 1) & 1;
        compute_tile(baseA + cur * ASZ, baseB + cur * BSZ);
    }
#endif

    // epilogue: alpha * acc + bias   (C layout: n = lane16, m = r + laneHi*8)
    const float alpha = *alpha_p;
#pragma unroll
    for (int mi = 0; mi < 4; ++mi) {
#pragma unroll
        for (int ni = 0; ni < 4; ++ni) {
            const int n  = nBase + nWave + ni * 16 + lane16;
            const float bv = bias[n];
            const int mrow = mBase + mWave + mi * 16 + laneHi * 8;
#pragma unroll
            for (int r = 0; r < 8; ++r)
                out[(size_t)(mrow + r) * NDIM + n] = alpha * acc[mi * 4 + ni][r] + bv;
        }
    }
}

// ---------------------------------------------------------------------------
extern "C" void kernel_launch(void* const* d_in, const int* in_sizes, int n_in,
                              void* d_out, int out_size, void* d_ws, size_t ws_size,
                              hipStream_t stream) {
    const float* x     = (const float*)d_in[0];   // [4,2048,2048]
    const float* w     = (const float*)d_in[1];   // [8192,2048]
    const float* alpha = (const float*)d_in[2];   // scalar
    const float* bias  = (const float*)d_in[3];   // [8192]
    float* out = (float*)d_out;                   // [4,2048,8192]

    char* ws = (char*)d_ws;
    float*          partials = (float*)ws;                          // 4096 f32
    float*          delta    = (float*)(ws + 16384);                // 1 f32
    unsigned short* xbf      = (unsigned short*)(ws + 32768);       // 32 MB
    unsigned short* wbf      = (unsigned short*)(ws + 32768 + XELEMS * 2);

    reduce_abs_sum<<<4096, 256, 0, stream>>>(w, partials);
    finalize_delta<<<1, 256, 0, stream>>>(partials, delta);
    quantize_w<<<(unsigned)(WELEMS / 4 / 256), 256, 0, stream>>>(w, delta, wbf);
    convert_x<<<(unsigned)(XELEMS / 4 / 256), 256, 0, stream>>>(x, xbf);

    dim3 grid(NDIM / 256, MDIM / 128);   // 32 x 64
    ternary_gemm_bf16<<<grid, 256, 0, stream>>>(xbf, wbf, alpha, bias, out);
}
